// MetaLA_5317169512706
// MI455X (gfx1250) — matrix operand order repaired and checked by reading, not compiled, not run on hardware
//
#include <hip/hip_runtime.h>

#define HID    1024
#define HEADS  8
#define HDIM   64
#define INNER  512
#define NQKV   1536
#define BATCH  4
#define SEQ    8192
#define MTOT   (BATCH*SEQ)   // 32768
#define CHUNK  128
#define NCHUNK (SEQ/CHUNK)   // 64
#define LN_EPS 1e-6f

typedef unsigned int   u32;
typedef unsigned short u16;
typedef _Float16 f16;
typedef f16    v16h  __attribute__((ext_vector_type(16)));
typedef __fp16 v2fp  __attribute__((ext_vector_type(2)));   // cvt_pkrtz return type
typedef float  v8f   __attribute__((ext_vector_type(8)));
typedef float  v4f   __attribute__((ext_vector_type(4)));
typedef u32    v4u   __attribute__((ext_vector_type(4)));

// ---- helpers -------------------------------------------------------------

__device__ inline v16h ld16h(const u16* p0, const u16* p1) {
    union { v4u v[2]; v16h h; } u;
    u.v[0] = *(const v4u*)p0;
    u.v[1] = *(const v4u*)p1;
    return u.h;
}

__device__ inline u32 pk2h(float a, float b) {
    v2fp p = __builtin_amdgcn_cvt_pkrtz(a, b);   // v_cvt_pk_f16_f32
    return __builtin_bit_cast(u32, p);
}

__device__ inline float sigm(float x) { return 1.0f / (1.0f + __expf(-x)); }

// Async global->LDS 16B copy (gfx1250 VFLAT GVS form), tracked by ASYNCcnt.
// lds_off: LDS byte offset (low 32 bits of a generic LDS pointer, per ISA
// aperture rules). voff: per-lane 32-bit byte offset. sbase: uniform base.
__device__ inline void async_ld_b128(u32 lds_off, u32 voff, const void* sbase) {
    asm volatile("global_load_async_to_lds_b128 %0, %1, %2"
                 :: "v"(lds_off), "v"(voff), "s"(sbase)
                 : "memory");
}
__device__ inline void wait_async0() {
    asm volatile("s_wait_asynccnt 0x0" ::: "memory");
}

// ---- kernel 1: weights -> f16, transposed [n][k] -------------------------

__global__ void __launch_bounds__(256)
k_cvt_weights(const float* Wq, const float* Wk, const float* Wv,
              const float* Wo, u16* wqkvT, u16* woT) {
    int idx = blockIdx.x * 256 + threadIdx.x;
    const int NW = NQKV * HID;
    if (idx < NW) {
        int n = idx / HID, k = idx % HID;
        float v;
        if      (n < INNER)     v = Wq[k * INNER + n];
        else if (n < 2 * INNER) v = Wk[k * INNER + (n - INNER)];
        else                    v = Wv[k * INNER + (n - 2 * INNER)];
        wqkvT[idx] = __builtin_bit_cast(u16, (f16)v);
    } else {
        int j = idx - NW;                     // WoT: [n=1024][k=512]
        int n = j / INNER, k = j % INNER;
        woT[j] = __builtin_bit_cast(u16, (f16)Wo[k * HID + n]);
    }
}

// ---- kernel 2: qkv = x @ WqkvT^T + b, f16 WMMA, fp32 out -----------------
// block: 256 threads (8 waves), tile 128x128, K-step 32.
// A tile: global fp32 -> f16 -> LDS (needs conversion, VGPR path).
// B tile: pure f16 copy -> async global->LDS (ASYNCcnt path).

__global__ void __launch_bounds__(256)
k_qkv_gemm(const float* x, const u16* wqkvT,
           const float* bq, const float* bk, const float* bv, float* qkv) {
    __shared__ u16 sA[128 * 32];   // [m][k] f16
    __shared__ u16 sB[128 * 32];   // [n][k] f16
    const int tid  = threadIdx.x;
    const int m0   = blockIdx.x * 128;
    const int n0   = blockIdx.y * 128;
    const int wave = tid >> 5, lane = tid & 31;
    const int half = lane >> 4, nl = lane & 15;
    const int wm = wave & 3, wn = wave >> 2;         // 4x2 wave grid

    v8f acc[2][4];
    v8f zero = {};
    for (int i = 0; i < 2; i++)
        for (int j = 0; j < 4; j++) acc[i][j] = zero;

    for (int kk = 0; kk < HID; kk += 32) {
        // stage B: async DMA of 128x32 f16 tile, [n][k]-contiguous in global
        #pragma unroll
        for (int r = 0; r < 2; r++) {
            int idx = tid + r * 256;
            int n = idx >> 2, kq = (idx & 3) * 8;
            u32 lds_off = (u32)(size_t)&sB[n * 32 + kq];
            u32 goff = (u32)(((size_t)(n0 + n) * HID + kk + kq) * sizeof(u16));
            async_ld_b128(lds_off, goff, wqkvT);
        }
        // stage A: 128x32 fp32 -> f16 LDS
        #pragma unroll
        for (int r = 0; r < 4; r++) {
            int idx = tid + r * 256;
            int m = idx >> 3, kq = (idx & 7) * 4;
            v4f g = *(const v4f*)&x[(size_t)(m0 + m) * HID + kk + kq];
            *(u32*)&sA[m * 32 + kq]     = pk2h(g.x, g.y);
            *(u32*)&sA[m * 32 + kq + 2] = pk2h(g.z, g.w);
        }
        if (kk + 32 < HID)
            __builtin_prefetch(&x[(size_t)(m0 + (tid >> 3)) * HID + kk + 32], 0, 1);
        wait_async0();
        __syncthreads();

        #pragma unroll
        for (int i = 0; i < 2; i++) {
            int row = (wm * 2 + i) * 16 + nl;
            v16h af = ld16h(&sA[row * 32 + half * 8], &sA[row * 32 + half * 8 + 16]);
            #pragma unroll
            for (int j = 0; j < 4; j++) {
                int n = wn * 64 + j * 16 + nl;
                v16h bf = ld16h(&sB[n * 32 + half * 8], &sB[n * 32 + half * 8 + 16]);
                acc[i][j] = __builtin_amdgcn_wmma_f32_16x16x32_f16(
                    false, af, false, bf, (short)0, acc[i][j], false, false);
            }
        }
        __syncthreads();
    }

    #pragma unroll
    for (int i = 0; i < 2; i++) {
        #pragma unroll
        for (int j = 0; j < 4; j++) {
            int col = n0 + wn * 64 + j * 16 + nl;
            float bias = (col < INNER) ? bq[col]
                       : (col < 2 * INNER) ? bk[col - INNER]
                       : bv[col - 2 * INNER];
            int mbase = m0 + (wm * 2 + i) * 16 + half * 8;
            #pragma unroll
            for (int t = 0; t < 8; t++)
                qkv[(size_t)(mbase + t) * NQKV + col] = acc[i][j][t] + bias;
        }
    }
}

// ---- kernel 3: attn = softmax_h(q.k / sqrt(d)), one wave per token -------

__global__ void __launch_bounds__(256)
k_attn(const float* qkv, float* attn) {
    int wave = threadIdx.x >> 5, lane = threadIdx.x & 31;
    int tok = blockIdx.x * 8 + wave;
    int h = lane >> 2, p = lane & 3;                  // 4 lanes per head
    const float* qr = qkv + (size_t)tok * NQKV + h * HDIM + p * 16;
    const float* kr = qr + INNER;
    float s = 0.f;
    #pragma unroll
    for (int i = 0; i < 16; i++) s += qr[i] * kr[i];
    s += __shfl_xor(s, 1, 32);
    s += __shfl_xor(s, 2, 32);                        // head dot on all 4 lanes
    float logit = s * 0.125f;                         // d^-0.5
    float mx = logit;
    mx = fmaxf(mx, __shfl_xor(mx, 4, 32));
    mx = fmaxf(mx, __shfl_xor(mx, 8, 32));
    mx = fmaxf(mx, __shfl_xor(mx, 16, 32));
    float e = __expf(logit - mx);
    float tot = e;
    tot += __shfl_xor(tot, 4, 32);                    // one lane per head in group
    tot += __shfl_xor(tot, 8, 32);
    tot += __shfl_xor(tot, 16, 32);
    if (p == 0) attn[(size_t)tok * HEADS + h] = e / tot;
}

// ---- kernel 4: per-chunk local scan (zero init) --------------------------

__global__ void __launch_bounds__(512)
k_scan_partial(const float* qkv, const float* attn, const float* alpha,
               float* chunkZ) {
    int b = blockIdx.x >> 6, c = blockIdx.x & 63;
    int hd = threadIdx.x, h = hd >> 6;
    float dec = sigm(alpha[hd]);
    float om = 1.f - dec;
    float st = 0.f;
    int tok0 = b * SEQ + c * CHUNK;
    for (int j = 0; j < CHUNK; j++) {
        int tok = tok0 + j;
        float a = attn[(size_t)tok * HEADS + h];
        float v = qkv[(size_t)tok * NQKV + 2 * INNER + hd];
        st = dec * st + om * a * v;
    }
    chunkZ[(size_t)(b * NCHUNK + c) * INNER + hd] = st;
}

// ---- kernel 5: prefix combine across chunks (64 serial steps) ------------

__global__ void __launch_bounds__(512)
k_scan_prefix(const float* chunkZ, const float* alpha, float* chunkI) {
    int b = blockIdx.x, hd = threadIdx.x;
    float dec = sigm(alpha[hd]);
    float dL = dec;
    #pragma unroll
    for (int i = 0; i < 7; i++) dL *= dL;             // dec^128
    float carry = 0.f;
    for (int c = 0; c < NCHUNK; c++) {
        size_t o = (size_t)(b * NCHUNK + c) * INNER + hd;
        chunkI[o] = carry;                            // state at chunk start
        carry = dL * carry + chunkZ[o];
    }
}

// ---- kernel 6: replay + core = q*state + sigmoid(q*aug), f16 out ---------

__global__ void __launch_bounds__(512)
k_scan_out(const float* qkv, const float* attn, const float* alpha,
           const float* aug_param, const float* chunkI, u16* core16) {
    int b = blockIdx.x >> 6, c = blockIdx.x & 63;
    int hd = threadIdx.x, h = hd >> 6;
    float dec = sigm(alpha[hd]);
    float om = 1.f - dec;
    float aug = aug_param[hd];
    float st = chunkI[(size_t)(b * NCHUNK + c) * INNER + hd];
    int tok0 = b * SEQ + c * CHUNK;
    for (int j = 0; j < CHUNK; j++) {
        int tok = tok0 + j;
        float a = attn[(size_t)tok * HEADS + h];
        const float* row = qkv + (size_t)tok * NQKV;
        float v = row[2 * INNER + hd];
        st = dec * st + om * a * v;
        float q = row[hd];
        float o = q * st + sigm(q * aug);
        core16[(size_t)tok * INNER + hd] = __builtin_bit_cast(u16, (f16)o);
    }
}

// ---- kernel 7: out = core @ WoT^T + bo, fused LayerNorm ------------------
// M-tile = 16 rows/block; each of 8 waves owns 128 output cols.

__global__ void __launch_bounds__(256)
k_out_gemm_ln(const u16* core16, const u16* woT, const float* bo,
              const float* ln_scale, const float* ln_bias, float* out) {
    __shared__ float sS1[16];
    __shared__ float sS2[16];
    const int tid = threadIdx.x;
    const int t0 = blockIdx.x * 16;
    const int wave = tid >> 5, lane = tid & 31;
    const int half = lane >> 4, nl = lane & 15;

    v8f acc[8];
    v8f zero = {};
    #pragma unroll
    for (int j = 0; j < 8; j++) acc[j] = zero;

    for (int kk = 0; kk < INNER; kk += 32) {
        const u16* ap = core16 + (size_t)(t0 + nl) * INNER + kk + half * 8;
        v16h af = ld16h(ap, ap + 16);
        if (kk + 32 < INNER)
            __builtin_prefetch(core16 + (size_t)(t0 + nl) * INNER + kk + 32, 0, 1);
        #pragma unroll
        for (int j = 0; j < 8; j++) {
            int n = wave * 128 + j * 16 + nl;
            const u16* bp = woT + (size_t)n * INNER + kk + half * 8;
            v16h bf = ld16h(bp, bp + 16);
            acc[j] = __builtin_amdgcn_wmma_f32_16x16x32_f16(
                false, af, false, bf, (short)0, acc[j], false, false);
        }
    }

    #pragma unroll
    for (int j = 0; j < 8; j++) {                     // + bo
        float bb = bo[wave * 128 + j * 16 + nl];
        #pragma unroll
        for (int t = 0; t < 8; t++) acc[j][t] += bb;
    }

    if (tid < 16) { sS1[tid] = 0.f; sS2[tid] = 0.f; }
    __syncthreads();
    #pragma unroll
    for (int t = 0; t < 8; t++) {                     // row stats via LDS atomics
        float v1 = 0.f, v2 = 0.f;
        #pragma unroll
        for (int j = 0; j < 8; j++) { float v = acc[j][t]; v1 += v; v2 += v * v; }
        v1 += __shfl_xor(v1, 1, 16);  v2 += __shfl_xor(v2, 1, 16);
        v1 += __shfl_xor(v1, 2, 16);  v2 += __shfl_xor(v2, 2, 16);
        v1 += __shfl_xor(v1, 4, 16);  v2 += __shfl_xor(v2, 4, 16);
        v1 += __shfl_xor(v1, 8, 16);  v2 += __shfl_xor(v2, 8, 16);
        if (nl == 0) {
            atomicAdd(&sS1[half * 8 + t], v1);        // ds_add_f32
            atomicAdd(&sS2[half * 8 + t], v2);
        }
    }
    __syncthreads();

    const float inv = 1.f / 1024.f;
    #pragma unroll
    for (int j = 0; j < 8; j++) {
        int col = wave * 128 + j * 16 + nl;
        float sc = ln_scale[col], bi = ln_bias[col];
        #pragma unroll
        for (int t = 0; t < 8; t++) {
            int m = half * 8 + t;
            float mean = sS1[m] * inv;
            float var  = sS2[m] * inv - mean * mean;
            float rstd = rsqrtf(var + LN_EPS);
            out[(size_t)(t0 + m) * HID + col] = (acc[j][t] - mean) * rstd * sc + bi;
        }
    }
}

// ---- host launcher -------------------------------------------------------

extern "C" void kernel_launch(void* const* d_in, const int* in_sizes, int n_in,
                              void* d_out, int out_size, void* d_ws, size_t ws_size,
                              hipStream_t stream) {
    const float* x     = (const float*)d_in[0];
    const float* Wq    = (const float*)d_in[1];
    const float* bq    = (const float*)d_in[2];
    const float* Wk    = (const float*)d_in[3];
    const float* bk    = (const float*)d_in[4];
    const float* Wv    = (const float*)d_in[5];
    const float* bv    = (const float*)d_in[6];
    const float* Wo    = (const float*)d_in[7];
    const float* bo    = (const float*)d_in[8];
    const float* alpha = (const float*)d_in[9];
    const float* aug   = (const float*)d_in[10];
    const float* lns   = (const float*)d_in[11];
    const float* lnb   = (const float*)d_in[12];
    float* out = (float*)d_out;

    char* ws = (char*)d_ws;
    u16*   wqkvT  = (u16*)(ws + 0);            // 1536*1024*2 = 3,145,728
    u16*   woT    = (u16*)(ws + 3145728);      // 1024*512*2  = 1,048,576
    float* qkv    = (float*)(ws + 4194304);    // 32768*1536*4 = 201,326,592
    float* attn   = (float*)(ws + 205520896);  // 32768*8*4    = 1,048,576
    float* chunkZ = (float*)(ws + 206569472);  // 4*64*512*4   = 524,288
    float* chunkI = (float*)(ws + 207093760);  // 524,288
    u16*   core16 = (u16*)(ws + 207618048);    // 32768*512*2  = 33,554,432

    k_cvt_weights<<<(NQKV * HID + HID * INNER) / 256, 256, 0, stream>>>(
        Wq, Wk, Wv, Wo, wqkvT, woT);
    k_qkv_gemm<<<dim3(MTOT / 128, NQKV / 128), 256, 0, stream>>>(
        x, wqkvT, bq, bk, bv, qkv);
    k_attn<<<MTOT / 8, 256, 0, stream>>>(qkv, attn);
    k_scan_partial<<<BATCH * NCHUNK, 512, 0, stream>>>(qkv, attn, alpha, chunkZ);
    k_scan_prefix<<<BATCH, 512, 0, stream>>>(chunkZ, alpha, chunkI);
    k_scan_out<<<BATCH * NCHUNK, 512, 0, stream>>>(qkv, attn, alpha, aug, chunkI, core16);
    k_out_gemm_ln<<<MTOT / 16, 256, 0, stream>>>(core16, woT, bo, lns, lnb, out);
}
